// OptimalTransport_87703232184354
// MI455X (gfx1250) — compile-verified, matrix operand order checked by reading, not run on hardware
//
#include <hip/hip_runtime.h>

// ---------------------------------------------------------------------------
// Sinkhorn optimal transport for MI455X (gfx1250, wave32, WMMA bf16)
//
//   K = exp(-C/0.05)  stored as bf16 in the LOWER HALF of d_out (256 MB)
//   10 iterations: fused row pass (u = 1/(Kv+eps) via v_wmma_f32_16x16x32_bf16)
//                  + column partials (K^T u, via v_fma_mix_f32_bf16) in the
//                  same K sweep, then a small deterministic reduce for v.
//   Final: out = u * exp(-C/0.05) * v  recomputed from C in fp32
//          (overwrites the staged K; reads only C / u / v -> no race).
//   d_ws only holds u and v (1 MB).
// ---------------------------------------------------------------------------

typedef __attribute__((ext_vector_type(16))) __bf16 v16bf;
typedef __attribute__((ext_vector_type(8)))  float  v8f;

#define B_DIM 128
#define N_DIM 1024
#define M_DIM 1024
#define NUM_ITERS 10
#define STAB   1e-8f
#define INVEPS 20.0f   // 1/0.05

union Frag16 { uint4 q[2]; v16bf v; };

__device__ __forceinline__ unsigned short f32_to_bf16_rne(float f) {
  unsigned int u = __float_as_uint(f);
  u += 0x7FFFu + ((u >> 16) & 1u);
  return (unsigned short)(u >> 16);
}

// acc += bf16(low half of packed) * ur   -- folds cvt into the FMA
__device__ __forceinline__ void fma_mix_lo(float& acc, unsigned int packed, float ur) {
  asm("v_fma_mix_f32_bf16 %0, %1, %2, %0 op_sel:[0,0,0] op_sel_hi:[1,0,0]"
      : "+v"(acc) : "v"(packed), "v"(ur));
}
// acc += bf16(high half of packed) * ur
__device__ __forceinline__ void fma_mix_hi(float& acc, unsigned int packed, float ur) {
  asm("v_fma_mix_f32_bf16 %0, %1, %2, %0 op_sel:[1,0,0] op_sel_hi:[1,0,0]"
      : "+v"(acc) : "v"(packed), "v"(ur));
}

// --------------------------- K = exp(-C/eps) -> bf16 -----------------------
__global__ __launch_bounds__(256) void gen_k_kernel(const float* __restrict__ C,
                                                    unsigned short* __restrict__ K) {
  size_t i = ((size_t)blockIdx.x * 256 + threadIdx.x) * 8;
  float4 c0 = *(const float4*)(C + i);
  float4 c1 = *(const float4*)(C + i + 4);
  uint4 o;
  o.x = (unsigned int)f32_to_bf16_rne(__expf(-INVEPS * c0.x)) |
        ((unsigned int)f32_to_bf16_rne(__expf(-INVEPS * c0.y)) << 16);
  o.y = (unsigned int)f32_to_bf16_rne(__expf(-INVEPS * c0.z)) |
        ((unsigned int)f32_to_bf16_rne(__expf(-INVEPS * c0.w)) << 16);
  o.z = (unsigned int)f32_to_bf16_rne(__expf(-INVEPS * c1.x)) |
        ((unsigned int)f32_to_bf16_rne(__expf(-INVEPS * c1.y)) << 16);
  o.w = (unsigned int)f32_to_bf16_rne(__expf(-INVEPS * c1.z)) |
        ((unsigned int)f32_to_bf16_rne(__expf(-INVEPS * c1.w)) << 16);
  *(uint4*)(K + i) = o;
}

// ------------------------------- v0 = 1/M ---------------------------------
__global__ __launch_bounds__(256) void init_v_kernel(float* __restrict__ v) {
  int i = blockIdx.x * 256 + threadIdx.x;
  v[i] = 1.0f / (float)M_DIM;
}

// ---------- fused pass: u = 1/(Kv+eps) ; vpart = per-block K^T u ----------
// grid = (N/64, B), block = 128 (4 waves, 16 rows each)
__global__ __launch_bounds__(128) void row_col_pass(
    const unsigned short* __restrict__ K,   // bf16 bits [B][N][M]
    const float* __restrict__ vin,          // [B][M]
    float* __restrict__ uout,               // [B][N]
    float* __restrict__ vpart)              // [B][16][M]
{
  __shared__ unsigned short vlds[M_DIM];
  __shared__ float colacc[4][M_DIM];

  const int b    = blockIdx.y;
  const int rb   = blockIdx.x;             // row block (64 rows)
  const int tid  = threadIdx.x;
  const int warp = tid >> 5;
  const int lane = tid & 31;
  const int half = lane >> 4;              // K-split half for WMMA frags

  // stage v as bf16 in LDS
  {
    const float* vb = vin + (size_t)b * M_DIM;
    for (int i = tid; i < M_DIM; i += 128)
      vlds[i] = f32_to_bf16_rne(vb[i]);
  }
  __syncthreads();

  const int rowbase = rb * 64 + warp * 16;
  const size_t kb   = (size_t)b * N_DIM * M_DIM;
  const unsigned short* rowPtr = K + kb + (size_t)(rowbase + (lane & 15)) * M_DIM;

  // ---- phase 1: row dots via WMMA bf16, B = v broadcast across columns ----
  v8f c0 = {0.f,0.f,0.f,0.f,0.f,0.f,0.f,0.f};
  v8f c1 = {0.f,0.f,0.f,0.f,0.f,0.f,0.f,0.f};
  #pragma unroll 4
  for (int m0 = 0; m0 < M_DIM; m0 += 64) {
    Frag16 a0, a1, b0, b1;
    a0.q[0] = *(const uint4*)(rowPtr + m0 +      half * 8);
    a0.q[1] = *(const uint4*)(rowPtr + m0 + 16 + half * 8);
    a1.q[0] = *(const uint4*)(rowPtr + m0 + 32 + half * 8);
    a1.q[1] = *(const uint4*)(rowPtr + m0 + 48 + half * 8);
    b0.q[0] = *(const uint4*)(vlds   + m0 +      half * 8);
    b0.q[1] = *(const uint4*)(vlds   + m0 + 16 + half * 8);
    b1.q[0] = *(const uint4*)(vlds   + m0 + 32 + half * 8);
    b1.q[1] = *(const uint4*)(vlds   + m0 + 48 + half * 8);
    c0 = __builtin_amdgcn_wmma_f32_16x16x32_bf16(false, a0.v, false, b0.v,
                                                 (short)0, c0, false, false);
    c1 = __builtin_amdgcn_wmma_f32_16x16x32_bf16(false, a1.v, false, b1.v,
                                                 (short)0, c1, false, false);
  }

  // lanes<16 hold row sums 0..7 (per C/D layout), lanes>=16 hold rows 8..15
  float u16v[16];
  #pragma unroll
  for (int r = 0; r < 8; ++r) {
    float own = c0[r] + c1[r];
    float oth = __shfl_xor(own, 16);
    float lo = (half == 0) ? own : oth;    // rows 0..7
    float hi = (half == 0) ? oth : own;    // rows 8..15
    u16v[r]     = 1.0f / (lo + STAB);
    u16v[8 + r] = 1.0f / (hi + STAB);
  }
  if ((lane & 15) == 0) {                  // lanes 0 and 16 write 8 rows each
    #pragma unroll
    for (int r = 0; r < 8; ++r)
      uout[(size_t)b * N_DIM + rowbase + half * 8 + r] = u16v[half * 8 + r];
  }

  // ---- phase 2: column partials K^T u over this warp's 16 rows (L2-hot) ---
  float acc2[32];
  #pragma unroll
  for (int i = 0; i < 32; ++i) acc2[i] = 0.f;
  const unsigned short* tbase = K + kb + (size_t)rowbase * M_DIM + lane * 8;
  for (int r = 0; r < 16; ++r) {
    const unsigned short* rp = tbase + (size_t)r * M_DIM;
    float ur = u16v[r];
    #pragma unroll
    for (int j = 0; j < 4; ++j) {
      uint4 q = *(const uint4*)(rp + 256 * j);
      unsigned int w[4] = {q.x, q.y, q.z, q.w};
      #pragma unroll
      for (int d = 0; d < 4; ++d) {
        fma_mix_lo(acc2[j * 8 + 2 * d],     w[d], ur);
        fma_mix_hi(acc2[j * 8 + 2 * d + 1], w[d], ur);
      }
    }
  }

  // deterministic block reduction of the 4 warps' partials
  #pragma unroll
  for (int j = 0; j < 4; ++j) {
    *(float4*)&colacc[warp][256 * j + lane * 8]     =
        make_float4(acc2[j*8+0], acc2[j*8+1], acc2[j*8+2], acc2[j*8+3]);
    *(float4*)&colacc[warp][256 * j + lane * 8 + 4] =
        make_float4(acc2[j*8+4], acc2[j*8+5], acc2[j*8+6], acc2[j*8+7]);
  }
  __syncthreads();
  {
    int col = tid * 8;
    float4 s0 = *(float4*)&colacc[0][col];
    float4 s1 = *(float4*)&colacc[0][col + 4];
    #pragma unroll
    for (int w = 1; w < 4; ++w) {
      float4 t0 = *(float4*)&colacc[w][col];
      float4 t1 = *(float4*)&colacc[w][col + 4];
      s0.x += t0.x; s0.y += t0.y; s0.z += t0.z; s0.w += t0.w;
      s1.x += t1.x; s1.y += t1.y; s1.z += t1.z; s1.w += t1.w;
    }
    float* dst = vpart + ((size_t)b * 16 + rb) * M_DIM + col;
    *(float4*)dst       = s0;
    *(float4*)(dst + 4) = s1;
  }
}

// ---------------- v = 1/(sum of 16 column partials + eps) ------------------
__global__ __launch_bounds__(256) void col_finalize(const float* __restrict__ vpart,
                                                    float* __restrict__ v) {
  int idx = blockIdx.x * 256 + threadIdx.x;        // over B*M
  int b = idx >> 10, m = idx & 1023;
  const float* p = vpart + (size_t)b * 16 * M_DIM + m;
  float s = 0.f;
  #pragma unroll
  for (int j = 0; j < 16; ++j) s += p[j * M_DIM];
  v[idx] = 1.0f / (s + STAB);
}

// ---------------- out = u * exp(-C/eps) * v  (fp32 K from C) ---------------
__global__ __launch_bounds__(256) void finalize_kernel(const float* __restrict__ C,
    const float* __restrict__ u, const float* __restrict__ v,
    float* __restrict__ out) {
  size_t i = ((size_t)blockIdx.x * 256 + threadIdx.x) * 4;
  float4 c = *(const float4*)(C + i);
  size_t bn = i >> 10;
  size_t b  = i >> 20;
  int    m  = (int)(i & 1023);
  float un  = u[bn];
  float4 vv = *(const float4*)(v + b * M_DIM + m);
  float4 o;
  o.x = un * __expf(-INVEPS * c.x) * vv.x;
  o.y = un * __expf(-INVEPS * c.y) * vv.y;
  o.z = un * __expf(-INVEPS * c.z) * vv.z;
  o.w = un * __expf(-INVEPS * c.w) * vv.w;
  *(float4*)(out + i) = o;
}

extern "C" void kernel_launch(void* const* d_in, const int* in_sizes, int n_in,
                              void* d_out, int out_size, void* d_ws, size_t ws_size,
                              hipStream_t stream) {
  (void)in_sizes; (void)n_in; (void)out_size; (void)ws_size;
  const float* C = (const float*)d_in[0];

  // bf16 K in lower 256 MB of d_out; column partials (8 MB) right after it.
  unsigned short* K = (unsigned short*)d_out;
  float* vpart = (float*)((char*)d_out + (size_t)B_DIM * N_DIM * M_DIM * 2);
  // u, v in workspace (1 MB total)
  float* u = (float*)d_ws;
  float* v = u + (size_t)B_DIM * N_DIM;

  const size_t total = (size_t)B_DIM * N_DIM * M_DIM;

  gen_k_kernel<<<(unsigned)(total / (256 * 8)), 256, 0, stream>>>(C, K);
  init_v_kernel<<<(B_DIM * M_DIM) / 256, 256, 0, stream>>>(v);

  dim3 grid(N_DIM / 64, B_DIM);
  for (int it = 0; it < NUM_ITERS; ++it) {
    row_col_pass<<<grid, 128, 0, stream>>>(K, v, u, vpart);
    col_finalize<<<(B_DIM * M_DIM) / 256, 256, 0, stream>>>(vpart, v);
  }

  finalize_kernel<<<(unsigned)(total / (256 * 4)), 256, 0, stream>>>(C, u, v, (float*)d_out);
}
